// MultiHeadAttention_6433861009947
// MI455X (gfx1250) — matrix operand order repaired and checked by reading, not compiled, not run on hardware
//
#include <hip/hip_runtime.h>
#include <hip/hip_bf16.h>

// MHA forward for B=8, T=1024, E=1024, H=16, D=64 on gfx1250 (wave32, WMMA).
#define BB 8
#define TT 1024
#define EE 1024
#define HH 16
#define DD 64

typedef __bf16 bf16_t;
typedef __attribute__((ext_vector_type(16))) __bf16 v16bf;
typedef __attribute__((ext_vector_type(8)))  float  v8f;

static __device__ __forceinline__ unsigned pk2(float lo, float hi) {
    union { __bf16 b[2]; unsigned u; } t;
    t.b[0] = (__bf16)lo; t.b[1] = (__bf16)hi;
    return t.u;
}

// 16 fp32 -> 16 bf16 packed into two uint4.
static __device__ __forceinline__ void pack16(const float* src, uint4& q0, uint4& q1) {
    const float4* s = (const float4*)src;
    float4 f0 = s[0], f1 = s[1], f2 = s[2], f3 = s[3];
    q0 = { pk2(f0.x,f0.y), pk2(f0.z,f0.w), pk2(f1.x,f1.y), pk2(f1.z,f1.w) };
    q1 = { pk2(f2.x,f2.y), pk2(f2.z,f2.w), pk2(f3.x,f3.y), pk2(f3.z,f3.w) };
}

// A-fragment for V_WMMA_*_16X16X32 (16x32 bf16), source row-major, ld elems.
// ISA layout: lane<16 holds row=lane, K{0..7,16..23}; lane>=16: K{8..15,24..31}.
static __device__ __forceinline__ v16bf load_a_frag(const bf16_t* base, int ld, int lane) {
    const int r  = lane & 15;
    const int kh = (lane >> 4) << 3;              // 0 or 8
    const bf16_t* p = base + (size_t)r * ld + kh;
    union { v16bf v; uint4 q[2]; } u;
    u.q[0] = *(const uint4*)(p);                  // K = kh+0..7
    u.q[1] = *(const uint4*)(p + 16);             // K = kh+16..23
    return u.v;
}

// B-fragment (32x16 bf16) with B[k][n] = Mem[n][k] (Mem row-major, ld elems).
// ISA layout: lane<16 holds column n=lane, K{0..7,16..23}; lane>=16: n=lane-16, K{8..15,24..31}.
static __device__ __forceinline__ v16bf load_bT_frag(const bf16_t* base, int ld, int lane) {
    const int n  = lane & 15;
    const int kh = (lane >> 4) << 3;              // 0 or 8
    const bf16_t* p = base + (size_t)n * ld + kh;
    union { v16bf v; uint4 q[2]; } u;
    u.q[0] = *(const uint4*)(p);                  // K = kh+0..7
    u.q[1] = *(const uint4*)(p + 16);             // K = kh+16..23
    return u.v;
}

static __device__ __forceinline__ v8f wmma_bf16(v16bf a, v16bf b, v8f c) {
    return __builtin_amdgcn_wmma_f32_16x16x32_bf16(
        /*neg_a=*/false, a, /*neg_b=*/false, b,
        /*c_mod=*/(short)0, c, /*reuse_a=*/false, /*reuse_b=*/false);
}

// ---------------------------------------------------------------------------
// Projection GEMM: Y = X @ W^T + bias  (M=8192, N=K=1024), bf16 WMMA, fp32 acc.
// Double-buffered LDS tiles with register staging: one barrier per K-step and
// next-tile global loads overlapped with the 8 WMMAs of the current tile.
// MODE 0: X fp32, write bf16 to (B,H,T,D)      (Q and K; Q folds 1/sqrt(D))
// MODE 1: X fp32, write bf16 to (B,H,D,T)      (V transposed for P@V B-frags)
// MODE 2: X bf16, write fp32 row-major (final output projection)
// ---------------------------------------------------------------------------
template <int MODE>
__global__ __launch_bounds__(256)
void proj_gemm(const float* __restrict__ Xf, const bf16_t* __restrict__ Xb,
               const float* __restrict__ W, const float* __restrict__ bias,
               bf16_t* __restrict__ outb, float* __restrict__ outf, float scale)
{
    constexpr int LDA = 40;                       // 32 + 8 pad (80B rows, 16B aligned)
    __shared__ __align__(16) bf16_t As[2][128 * LDA];
    __shared__ __align__(16) bf16_t Bs[2][128 * LDA];

    const int tid  = threadIdx.x;
    const int lane = tid & 31;
    const int wave = tid >> 5;
    const int wm   = wave & 3;                    // 4 row strips of 32
    const int wn   = wave >> 2;                   // 2 col strips of 64
    const int m0   = blockIdx.y * 128;
    const int n0   = blockIdx.x * 128;

    v8f acc[2][4];
    #pragma unroll
    for (int mi = 0; mi < 2; ++mi)
        #pragma unroll
        for (int ni = 0; ni < 4; ++ni)
            #pragma unroll
            for (int j = 0; j < 8; ++j) acc[mi][ni][j] = 0.0f;

    const int lrow = tid >> 1;                    // 0..127
    const int lcol = (tid & 1) << 4;              // 0 or 16

    // stage one 128x32 X-tile and one 128x32 W-tile (as packed bf16) into regs
    uint4 ra0, ra1, rb0, rb1;
    auto stage = [&](int k0) {
        if constexpr (MODE == 2) {
            const bf16_t* src = Xb + (size_t)(m0 + lrow) * EE + k0 + lcol;
            ra0 = *(const uint4*)(src);
            ra1 = *(const uint4*)(src + 8);
        } else {
            pack16(Xf + (size_t)(m0 + lrow) * EE + k0 + lcol, ra0, ra1);
        }
        pack16(W + (size_t)(n0 + lrow) * EE + k0 + lcol, rb0, rb1);
    };

    stage(0);
    int p = 0;
    for (int k0 = 0; k0 < EE; k0 += 32, p ^= 1) {
        // commit staged tile to LDS buffer p
        *(uint4*)&As[p][lrow * LDA + lcol]     = ra0;
        *(uint4*)&As[p][lrow * LDA + lcol + 8] = ra1;
        *(uint4*)&Bs[p][lrow * LDA + lcol]     = rb0;
        *(uint4*)&Bs[p][lrow * LDA + lcol + 8] = rb1;
        __syncthreads();

        // start next tile's global loads (overlap with WMMAs below)
        if (k0 + 32 < EE) stage(k0 + 32);

        v16bf af[2], bfr[4];
        #pragma unroll
        for (int mi = 0; mi < 2; ++mi)
            af[mi] = load_a_frag(&As[p][(wm * 32 + mi * 16) * LDA], LDA, lane);
        #pragma unroll
        for (int ni = 0; ni < 4; ++ni)
            bfr[ni] = load_bT_frag(&Bs[p][(wn * 64 + ni * 16) * LDA], LDA, lane);
        #pragma unroll
        for (int mi = 0; mi < 2; ++mi)
            #pragma unroll
            for (int ni = 0; ni < 4; ++ni)
                acc[mi][ni] = wmma_bf16(af[mi], bfr[ni], acc[mi][ni]);
        // no trailing barrier: next iteration writes the other buffer, and its
        // __syncthreads() orders those stores against this iteration's reads
    }

    // epilogue: C-layout lane L, vgpr j -> row = j + 8*(L>=16), col = L&15
    #pragma unroll
    for (int mi = 0; mi < 2; ++mi) {
        #pragma unroll
        for (int ni = 0; ni < 4; ++ni) {
            const int mrow0 = m0 + wm * 32 + mi * 16 + ((lane >> 4) << 3);
            const int ncol  = n0 + wn * 64 + ni * 16 + (lane & 15);
            const float bn  = bias[ncol];
            #pragma unroll
            for (int j = 0; j < 8; ++j) {
                const int m = mrow0 + j;
                const float y = (acc[mi][ni][j] + bn) * scale;
                if constexpr (MODE == 0) {
                    const int b = m >> 10, t = m & 1023, h = ncol >> 6, d = ncol & 63;
                    outb[((size_t)(b * HH + h) * TT + t) * DD + d] = (bf16_t)y;
                } else if constexpr (MODE == 1) {
                    const int b = m >> 10, t = m & 1023, h = ncol >> 6, d = ncol & 63;
                    outb[((size_t)(b * HH + h) * DD + d) * TT + t] = (bf16_t)y;
                } else {
                    outf[(size_t)m * EE + ncol] = y;
                }
            }
        }
    }
}

// ---------------------------------------------------------------------------
// Flash-attention forward: one block = 128 query rows of one (b,h).
// Q,K in (B,H,T,D) bf16; Vt in (B,H,D,T) bf16; Aout bf16 row-major (B*T, E).
// Each wave owns 16 query rows; streams key tiles of 64 with online softmax.
// ---------------------------------------------------------------------------
__global__ __launch_bounds__(256)
void attn_fwd(const bf16_t* __restrict__ Q, const bf16_t* __restrict__ K,
              const bf16_t* __restrict__ Vt, bf16_t* __restrict__ Aout)
{
    __shared__ __align__(16) bf16_t plds[8][16][64];   // per-wave P repack scratch

    const int lane = threadIdx.x & 31;
    const int wave = threadIdx.x >> 5;
    const int bh   = blockIdx.x;                       // b*H + h
    const int b    = bh >> 4, h = bh & 15;
    const int qrow = blockIdx.y * 128 + wave * 16;

    const bf16_t* Qb = Q + ((size_t)bh * TT + qrow) * DD;
    const v16bf aq0 = load_a_frag(Qb,      DD, lane);  // K = d 0..31
    const v16bf aq1 = load_a_frag(Qb + 32, DD, lane);  // K = d 32..63

    v8f o[4];
    float rm[8], rl[8];
    #pragma unroll
    for (int nd = 0; nd < 4; ++nd)
        #pragma unroll
        for (int j = 0; j < 8; ++j) o[nd][j] = 0.0f;
    #pragma unroll
    for (int j = 0; j < 8; ++j) { rm[j] = -1e30f; rl[j] = 0.0f; }

    const bf16_t* Kb0 = K  + (size_t)bh * TT * DD;
    const bf16_t* Vb0 = Vt + (size_t)bh * DD * TT;

    for (int t0 = 0; t0 < TT; t0 += 64) {
        // prefetch next K/V tiles into cache while we compute on this one
        // (global_prefetch_b8: 8KB K tile + 64 rows x 128B of Vt columns)
        if (t0 + 64 < TT) {
            const bf16_t* kn = Kb0 + (size_t)(t0 + 64) * DD;
            __builtin_prefetch(kn + (size_t)lane * 128,      0, 3);
            __builtin_prefetch(kn + (size_t)lane * 128 + 64, 0, 3);
            __builtin_prefetch(Vb0 + (size_t)lane * TT        + t0 + 64, 0, 3);
            __builtin_prefetch(Vb0 + (size_t)(lane + 32) * TT + t0 + 64, 0, 3);
        }

        // S = Q K^T  (scale already folded into Q projection)
        v8f s[4];
        #pragma unroll
        for (int nt = 0; nt < 4; ++nt)
            #pragma unroll
            for (int j = 0; j < 8; ++j) s[nt][j] = 0.0f;
        #pragma unroll
        for (int nt = 0; nt < 4; ++nt) {
            const bf16_t* Kb = Kb0 + (size_t)(t0 + nt * 16) * DD;
            v16bf bk0 = load_bT_frag(Kb,      DD, lane);
            v16bf bk1 = load_bT_frag(Kb + 32, DD, lane);
            s[nt] = wmma_bf16(aq0, bk0, s[nt]);
            s[nt] = wmma_bf16(aq1, bk1, s[nt]);
        }

        // online softmax: row = j + 8*(lane>=16); row lives across 16 lanes of a half
        float mnew[8];
        #pragma unroll
        for (int j = 0; j < 8; ++j)
            mnew[j] = fmaxf(fmaxf(s[0][j], s[1][j]), fmaxf(s[2][j], s[3][j]));
        #pragma unroll
        for (int off = 1; off < 16; off <<= 1)
            #pragma unroll
            for (int j = 0; j < 8; ++j)
                mnew[j] = fmaxf(mnew[j], __shfl_xor(mnew[j], off, 32));
        float alpha[8], ps[8];
        #pragma unroll
        for (int j = 0; j < 8; ++j) {
            mnew[j]  = fmaxf(mnew[j], rm[j]);
            alpha[j] = __expf(rm[j] - mnew[j]);
            ps[j]    = 0.0f;
        }
        const int prow = (lane >> 4) << 3;
        const int pcol = lane & 15;
        #pragma unroll
        for (int nt = 0; nt < 4; ++nt)
            #pragma unroll
            for (int j = 0; j < 8; ++j) {
                const float pv = __expf(s[nt][j] - mnew[j]);
                ps[j] += pv;
                plds[wave][prow + j][nt * 16 + pcol] = (bf16_t)pv;  // C-layout -> LDS
            }
        #pragma unroll
        for (int off = 1; off < 16; off <<= 1)
            #pragma unroll
            for (int j = 0; j < 8; ++j)
                ps[j] += __shfl_xor(ps[j], off, 32);
        #pragma unroll
        for (int j = 0; j < 8; ++j) {
            rl[j] = rl[j] * alpha[j] + ps[j];
            rm[j] = mnew[j];
        }
        #pragma unroll
        for (int nd = 0; nd < 4; ++nd)
            #pragma unroll
            for (int j = 0; j < 8; ++j) o[nd][j] *= alpha[j];

        // LDS in same wave is in-order; wait + compiler fence before A-layout reads
        asm volatile("s_wait_dscnt 0" ::: "memory");
        v16bf pa0 = load_a_frag(&plds[wave][0][0],      64, lane);  // keys t0+0..31
        v16bf pa1 = load_a_frag(&plds[wave][0][0] + 32, 64, lane);  // keys t0+32..63

        // O += P @ V  (Vt rows are d, columns are t -> contiguous B-frags)
        #pragma unroll
        for (int nd = 0; nd < 4; ++nd) {
            const bf16_t* Vb = Vb0 + (size_t)(nd * 16) * TT + t0;
            v16bf bv0 = load_bT_frag(Vb,      TT, lane);
            v16bf bv1 = load_bT_frag(Vb + 32, TT, lane);
            o[nd] = wmma_bf16(pa0, bv0, o[nd]);
            o[nd] = wmma_bf16(pa1, bv1, o[nd]);
        }
        asm volatile("" ::: "memory");   // keep this iter's LDS reads before next stores
    }

    // normalize and write (B*T, E) bf16 for the output projection
    #pragma unroll
    for (int nd = 0; nd < 4; ++nd) {
        #pragma unroll
        for (int j = 0; j < 8; ++j) {
            const int t = qrow + j + ((lane >> 4) << 3);
            const int e = h * DD + nd * 16 + (lane & 15);
            const float y = o[nd][j] / rl[j];
            Aout[((size_t)b * TT + t) * EE + e] = (bf16_t)y;
        }
    }
}

// ---------------------------------------------------------------------------
extern "C" void kernel_launch(void* const* d_in, const int* in_sizes, int n_in,
                              void* d_out, int out_size, void* d_ws, size_t ws_size,
                              hipStream_t stream)
{
    const float* query = (const float*)d_in[0];
    const float* key   = (const float*)d_in[1];
    const float* value = (const float*)d_in[2];
    const float* Wq = (const float*)d_in[3];
    const float* bq = (const float*)d_in[4];
    const float* Wk = (const float*)d_in[5];
    const float* bk = (const float*)d_in[6];
    const float* Wv = (const float*)d_in[7];
    const float* bv = (const float*)d_in[8];
    const float* Wo = (const float*)d_in[9];
    const float* bo = (const float*)d_in[10];
    float* out = (float*)d_out;

    const size_t nElem = (size_t)BB * TT * EE;     // 8,388,608 elems (16 MB bf16)
    bf16_t* Qb = (bf16_t*)d_ws;
    bf16_t* Kb = Qb + nElem;
    bf16_t* Vt = Kb + nElem;
    bf16_t* At = Vt + nElem;                       // total 64 MB workspace

    const dim3 gg(EE / 128, (BB * TT) / 128);      // (8, 64)
    const dim3 bb(256);
    const float qscale = 0.125f;                   // 1/sqrt(D), folded into Q

    hipLaunchKernelGGL((proj_gemm<0>), gg, bb, 0, stream, query, (const bf16_t*)nullptr, Wq, bq, Qb, (float*)nullptr, qscale);
    hipLaunchKernelGGL((proj_gemm<0>), gg, bb, 0, stream, key,   (const bf16_t*)nullptr, Wk, bk, Kb, (float*)nullptr, 1.0f);
    hipLaunchKernelGGL((proj_gemm<1>), gg, bb, 0, stream, value, (const bf16_t*)nullptr, Wv, bv, Vt, (float*)nullptr, 1.0f);
    hipLaunchKernelGGL(attn_fwd, dim3(BB * HH, TT / 128), bb, 0, stream, Qb, Kb, Vt, At);
    hipLaunchKernelGGL((proj_gemm<2>), gg, bb, 0, stream, (const float*)nullptr, At, Wo, bo, (bf16_t*)nullptr, out, 1.0f);
}
